// GPRGNN_9345848836276
// MI455X (gfx1250) — matrix operand order repaired and checked by reading, not compiled, not run on hardware
//
#include <hip/hip_runtime.h>
#include <hip/hip_bf16.h>

typedef __attribute__((ext_vector_type(16))) __bf16 v16bf;
typedef __attribute__((ext_vector_type(8)))  float  v8f;
typedef __attribute__((ext_vector_type(4)))  float  v4f;

#define IN_FEATS  512
#define N_HIDDEN   64
#define N_CLASSES  16

union V16 {
    v16bf v;
    uint4 q[2];
};

__device__ __forceinline__ v8f v8f_zero() {
    v8f z;
#pragma unroll
    for (int j = 0; j < 8; ++j) z[j] = 0.0f;
    return z;
}

// ---------------------------------------------------------------------------
// Kernel 0: build bf16 transposed weights in workspace.
//   w1t[n*512 + k] = bf16(W1[k*64 + n])   (64 x 512)
//   w2t[n*64  + k] = bf16(W2[k*16 + n])   (16 x 64)
// ---------------------------------------------------------------------------
__global__ void gpr_convert_weights(const float* __restrict__ W1,
                                    const float* __restrict__ W2,
                                    __bf16* __restrict__ w1t,
                                    __bf16* __restrict__ w2t) {
    int tid = blockIdx.x * blockDim.x + threadIdx.x;
    if (tid < IN_FEATS * N_HIDDEN) {
        int k = tid / N_HIDDEN, n = tid % N_HIDDEN;
        w1t[n * IN_FEATS + k] = (__bf16)W1[tid];
    } else {
        int t = tid - IN_FEATS * N_HIDDEN;
        if (t < N_HIDDEN * N_CLASSES) {
            int k = t / N_CLASSES, n = t % N_CLASSES;
            w2t[n * N_HIDDEN + k] = (__bf16)W2[t];
        }
    }
}

// ---------------------------------------------------------------------------
// Kernel 1: fused MLP encoder via WMMA (bf16 inputs, f32 accumulate).
//   h = relu(feature @ W1 + b1); z = h @ W2 + b2
//   writes: z0 = z, hidden = temp[0]*z, z1 = 0 (ping-pong buffer init)
// One wave computes a 16-row tile; block of 8 waves covers 128 rows.
// ---------------------------------------------------------------------------
__global__ __launch_bounds__(256)
void gpr_mlp_wmma(const float* __restrict__ feat,
                  const __bf16* __restrict__ w1t,
                  const __bf16* __restrict__ w2t,
                  const float* __restrict__ b1,
                  const float* __restrict__ b2,
                  const float* __restrict__ temp,
                  float* __restrict__ z0,
                  float* __restrict__ z1,
                  float* __restrict__ hidden,
                  int nNodes) {
    // per-wave h-tile scratch; row stride 80 bf16 = 160B (16B aligned,
    // spreads banks for the ds_load_b128 transposed reads)
    __shared__ __bf16 h_lds[8][16][80];

    const int wave = threadIdx.x >> 5;
    const int lane = threadIdx.x & 31;
    const int half = lane >> 4;       // lane-half select (ISA A/B layouts)
    const int l16  = lane & 15;
    const int row0 = blockIdx.x * 128 + wave * 16;

    v8f acc[4];
#pragma unroll
    for (int i = 0; i < 4; ++i) acc[i] = v8f_zero();

    // A-matrix row for this lane (M = l16), clamped for the ragged tail
    int ar = row0 + l16;
    if (ar >= nNodes) ar = nNodes - 1;
    const float* aptr = feat + (size_t)ar * IN_FEATS;

    // ---- GEMM1: 16 K-steps of 32, 4 N-tiles of 16 -> 64 WMMAs / wave ----
    for (int k0 = 0; k0 < IN_FEATS; k0 += 32) {
        // A 16x32 bf16 layout: element e = g*8+w  ->  k = k0 + g*16 + half*8 + w
        V16 a;
#pragma unroll
        for (int g = 0; g < 2; ++g) {
            const int kb = k0 + g * 16 + half * 8;
            // features are streamed exactly once -> non-temporal loads keep
            // L2 free for W1t / z buffers used by the propagation phase
            const v4f f0 = __builtin_nontemporal_load((const v4f*)(aptr + kb));
            const v4f f1 = __builtin_nontemporal_load((const v4f*)(aptr + kb + 4));
#pragma unroll
            for (int j = 0; j < 4; ++j) {
                a.v[g * 8 + j]     = (__bf16)f0[j];   // native cvt_pk_bf16_f32
                a.v[g * 8 + 4 + j] = (__bf16)f1[j];
            }
        }
#pragma unroll
        for (int nt = 0; nt < 4; ++nt) {
            // B 32x16 bf16 layout: lane n = l16; element e = g*8+w -> k = k0 + g*16 + half*8 + w
            const __bf16* bp =
                w1t + (size_t)(nt * 16 + l16) * IN_FEATS + k0 + half * 8;
            V16 b;
            b.q[0] = *(const uint4*)(bp);
            b.q[1] = *(const uint4*)(bp + 16);
            acc[nt] = __builtin_amdgcn_wmma_f32_16x16x32_bf16(
                false, a.v, false, b.v, (short)0, acc[nt], false, false);
        }
    }

    // ---- epilogue 1: +b1, relu, transpose C-layout -> row-major bf16 in LDS
#pragma unroll
    for (int nt = 0; nt < 4; ++nt) {
        const int n_g = nt * 16 + l16;
        const float bias = b1[n_g];
#pragma unroll
        for (int v = 0; v < 8; ++v) {
            float h = acc[nt][v] + bias;
            h = h > 0.0f ? h : 0.0f;
            const int m = v + 8 * half;          // C/D layout: M = v + 8*half
            h_lds[wave][m][n_g] = (__bf16)h;
        }
    }
    __syncthreads();

    // ---- GEMM2: z[16x16] = h[16x64] @ W2[64x16], 2 WMMAs ----
    v8f acc2 = v8f_zero();
#pragma unroll
    for (int kt = 0; kt < 2; ++kt) {
        V16 a2;
#pragma unroll
        for (int g = 0; g < 2; ++g) {
            const int kb = kt * 32 + g * 16 + half * 8;   // A layout, M = l16
            a2.q[g] = *(const uint4*)(&h_lds[wave][l16][kb]);
        }
        const __bf16* bp = w2t + l16 * N_HIDDEN + kt * 32 + half * 8;
        V16 b;
        b.q[0] = *(const uint4*)(bp);
        b.q[1] = *(const uint4*)(bp + 16);
        acc2 = __builtin_amdgcn_wmma_f32_16x16x32_bf16(
            false, a2.v, false, b.v, (short)0, acc2, false, false);
    }

    // ---- epilogue 2: +b2; z0 = z, hidden = temp[0]*z, z1 = 0 ----
    const float t0 = temp[0];
    const float bias2 = b2[l16];
#pragma unroll
    for (int v = 0; v < 8; ++v) {
        const int m = v + 8 * half;
        const int row = row0 + m;
        if (row < nNodes) {
            const float zv = acc2[v] + bias2;
            const size_t idx = (size_t)row * N_CLASSES + l16;
            z0[idx] = zv;
            hidden[idx] = t0 * zv;
            z1[idx] = 0.0f;
        }
    }
}

// ---------------------------------------------------------------------------
// Kernel 2: SpMM hop. One thread per edge:
//   zout[dst, :] += w * zin[src, :]  (16 f32 atomics; z buffers are L2-resident)
// ---------------------------------------------------------------------------
__global__ __launch_bounds__(256)
void gpr_spmm(const long long* __restrict__ esrc,
              const long long* __restrict__ edst,
              const float* __restrict__ w,
              const float* __restrict__ zin,
              float* __restrict__ zout,
              int nE) {
    const int e = blockIdx.x * blockDim.x + threadIdx.x;
    if (e >= nE) return;

    // prefetch a future edge's gather row (global_prefetch_b8)
    const int ep = e + 256 * 64;
    if (ep < nE) {
        const long long sp = esrc[ep];
        __builtin_prefetch((const void*)(zin + sp * N_CLASSES), 0, 0);
    }

    const long long s = esrc[e];
    const long long d = edst[e];
    const float ww = w[e];
    const float* zi = zin + s * N_CLASSES;
    float* zo = zout + d * N_CLASSES;

    float4 x0 = *(const float4*)(zi);
    float4 x1 = *(const float4*)(zi + 4);
    float4 x2 = *(const float4*)(zi + 8);
    float4 x3 = *(const float4*)(zi + 12);

    atomicAdd(zo + 0,  ww * x0.x); atomicAdd(zo + 1,  ww * x0.y);
    atomicAdd(zo + 2,  ww * x0.z); atomicAdd(zo + 3,  ww * x0.w);
    atomicAdd(zo + 4,  ww * x1.x); atomicAdd(zo + 5,  ww * x1.y);
    atomicAdd(zo + 6,  ww * x1.z); atomicAdd(zo + 7,  ww * x1.w);
    atomicAdd(zo + 8,  ww * x2.x); atomicAdd(zo + 9,  ww * x2.y);
    atomicAdd(zo + 10, ww * x2.z); atomicAdd(zo + 11, ww * x2.w);
    atomicAdd(zo + 12, ww * x3.x); atomicAdd(zo + 13, ww * x3.y);
    atomicAdd(zo + 14, ww * x3.z); atomicAdd(zo + 15, ww * x3.w);
}

// ---------------------------------------------------------------------------
// Kernel 3: hidden += temp[k] * znext; zero the other ping-pong buffer.
// ---------------------------------------------------------------------------
__global__ __launch_bounds__(256)
void gpr_accum_clear(float* __restrict__ hidden,
                     const float* __restrict__ znext,
                     float* __restrict__ zclear,
                     const float* __restrict__ temp,
                     int k, int n) {
    const int i = blockIdx.x * blockDim.x + threadIdx.x;
    if (i < n) {
        hidden[i] += temp[k] * znext[i];
        zclear[i] = 0.0f;
    }
}

// ---------------------------------------------------------------------------
extern "C" void kernel_launch(void* const* d_in, const int* in_sizes, int n_in,
                              void* d_out, int out_size, void* d_ws, size_t ws_size,
                              hipStream_t stream) {
    const float*     feat  = (const float*)d_in[0];
    const float*     W1    = (const float*)d_in[1];
    const float*     b1    = (const float*)d_in[2];
    const float*     W2    = (const float*)d_in[3];
    const float*     b2    = (const float*)d_in[4];
    const long long* eidx  = (const long long*)d_in[5];   // int64 [2, nE]
    const float*     normA = (const float*)d_in[6];
    const float*     temp  = (const float*)d_in[7];

    const int nNodes = in_sizes[0] / IN_FEATS;            // 100000
    const int nE     = in_sizes[6];                       // 3200000
    const int KHOPS  = in_sizes[7] - 1;                   // 10

    // workspace layout
    float* zb0 = (float*)d_ws;
    float* zb1 = zb0 + (size_t)nNodes * N_CLASSES;
    __bf16* w1t = (__bf16*)(zb1 + (size_t)nNodes * N_CLASSES);
    __bf16* w2t = w1t + (size_t)N_HIDDEN * IN_FEATS;

    float* hidden = (float*)d_out;

    const int cvt_elems = IN_FEATS * N_HIDDEN + N_HIDDEN * N_CLASSES;
    gpr_convert_weights<<<(cvt_elems + 255) / 256, 256, 0, stream>>>(W1, W2, w1t, w2t);

    const int mblocks = (nNodes + 127) / 128;
    gpr_mlp_wmma<<<mblocks, 256, 0, stream>>>(feat, w1t, w2t, b1, b2, temp,
                                              zb0, zb1, hidden, nNodes);

    float* zcur  = zb0;
    float* znext = zb1;   // zeroed by gpr_mlp_wmma
    const int eblocks = (nE + 255) / 256;
    const int nblocks = ((nNodes * N_CLASSES) + 255) / 256;

    for (int k = 1; k <= KHOPS; ++k) {
        gpr_spmm<<<eblocks, 256, 0, stream>>>(eidx, eidx + nE, normA,
                                              zcur, znext, nE);
        gpr_accum_clear<<<nblocks, 256, 0, stream>>>(hidden, znext, zcur,
                                                     temp, k, nNodes * N_CLASSES);
        float* t = zcur; zcur = znext; znext = t;
    }
}
